// Block_68753836474893
// MI455X (gfx1250) — matrix operand order repaired and checked by reading, not compiled, not run on hardware
//
#include <hip/hip_runtime.h>
#include <math.h>

// ---------------------------------------------------------------------------
// CSWin-style block for MI455X (gfx1250).
// All GEMMs + flash attention via v_wmma_f32_16x16x32_bf16 (fp32 accum).
// Weights pre-packed to bf16 (N-major, K contiguous); K/V packed per (b,head):
// every WMMA fragment is a pair of contiguous 16B loads. GEMM stages its whole
// 16xK A-strip in LDS once (one barrier), then runs a barrier-free K loop with
// all four B fragments in flight before the WMMA group.
// fp32 VALU for LN / softmax / dwconv. Workspace ~87 MB.
// ---------------------------------------------------------------------------

#define BATCH 2
#define HH 48
#define WW 48
#define CC 256
#define NTOK (HH * WW)          // 2304
#define BN (BATCH * NTOK)       // 4608
#define HID (4 * CC)            // 1024
#define NHEAD 8
#define DH 32

typedef __attribute__((ext_vector_type(16))) __bf16 bf16x16;
typedef __attribute__((ext_vector_type(8)))  float  v8f;

__device__ __forceinline__ float gelu_exact(float x) {
    return 0.5f * x * (1.0f + erff(x * 0.70710678118654752440f));
}

// ------------------------------- LayerNorm ---------------------------------
__global__ __launch_bounds__(256) void ln_kernel(const float* __restrict__ in,
                                                 const float* __restrict__ g,
                                                 const float* __restrict__ beta,
                                                 float* __restrict__ out) {
    __shared__ float ws1[8], ws2[8];
    const int row = blockIdx.x;
    const int t = threadIdx.x;
    const float x = in[(size_t)row * CC + t];
    float s = x, s2 = x * x;
#pragma unroll
    for (int mask = 16; mask >= 1; mask >>= 1) {
        s  += __shfl_xor(s,  mask, 32);
        s2 += __shfl_xor(s2, mask, 32);
    }
    if ((t & 31) == 0) { ws1[t >> 5] = s; ws2[t >> 5] = s2; }
    __syncthreads();
    float ts = 0.f, ts2 = 0.f;
#pragma unroll
    for (int i = 0; i < 8; ++i) { ts += ws1[i]; ts2 += ws2[i]; }
    const float mu  = ts * (1.0f / CC);
    const float var = ts2 * (1.0f / CC) - mu * mu;
    const float rs  = rsqrtf(var + 1e-6f);
    out[(size_t)row * CC + t] = (x - mu) * rs * g[t] + beta[t];
}

// --------------------------- Depthwise conv (SAME) -------------------------
__global__ __launch_bounds__(256) void dwconv_kernel(const float* __restrict__ in,
                                                     const float* __restrict__ w,
                                                     const float* __restrict__ bias,
                                                     const float* __restrict__ resid,
                                                     float* __restrict__ out, int T) {
    const int idx = blockIdx.x * 256 + threadIdx.x;   // over B*H*W*C
    const int c = idx & (CC - 1);
    int sp = idx >> 8;
    const int x = sp % WW; sp /= WW;
    const int y = sp % HH;
    const int b = sp / HH;
    const int R = T >> 1;
    float acc = bias[c];
    for (int dy = 0; dy < T; ++dy) {
        const int yy = y + dy - R;
        if (yy < 0 || yy >= HH) continue;
        for (int dx = 0; dx < T; ++dx) {
            const int xx = x + dx - R;
            if (xx < 0 || xx >= WW) continue;
            acc += in[((size_t)((b * HH + yy) * WW + xx)) * CC + c] *
                   w[(dy * T + dx) * CC + c];
        }
    }
    if (resid) acc += resid[idx];
    out[idx] = acc;
}

// ------------------------------ Elementwise --------------------------------
__global__ __launch_bounds__(256) void add2_kernel(const float* __restrict__ a,
                                                   const float* __restrict__ b,
                                                   float* __restrict__ out) {
    const int i = blockIdx.x * 256 + threadIdx.x;
    out[i] = a[i] + b[i];
}

__global__ __launch_bounds__(256) void combine_kernel(const float* __restrict__ x,
                                                      const float* __restrict__ u,
                                                      const float* __restrict__ v,
                                                      float* __restrict__ out) {
    const int i = blockIdx.x * 256 + threadIdx.x;
    out[i] = x[i] + u[i] * v[i];
}

// --------------------------- Weight pack (fp32 -> bf16) --------------------
// W[k*N + n] -> Wp[n*K + k]  (N-major, K contiguous: B-fragment friendly)
__global__ __launch_bounds__(256) void packw_kernel(const float* __restrict__ W,
                                                    __bf16* __restrict__ Wp,
                                                    int K, int N) {
    const int idx = blockIdx.x * 256 + threadIdx.x;   // over K*N
    const int n = idx % N;
    const int k = idx / N;
    Wp[(size_t)n * K + k] = (__bf16)W[(size_t)k * N + n];
}

// ------------------- K/V pack for attention (fp32 -> bf16) -----------------
// qkv: (B*N, 768).  Kp[(b,h,key),d] (d contiguous), Vt[(b,h,d),key] (key contig)
__global__ __launch_bounds__(256) void packkv_kernel(const float* __restrict__ qkv,
                                                     __bf16* __restrict__ Kp,
                                                     __bf16* __restrict__ Vt) {
    const int idx = blockIdx.x * 256 + threadIdx.x;   // over BN*CC
    const int c = idx & (CC - 1);
    const int row = idx >> 8;                          // global token row
    const int b = row / NTOK;
    const int t = row - b * NTOK;
    const int h = c >> 5;
    const int d = c & (DH - 1);
    const int bh = b * NHEAD + h;
    const float kv = qkv[(size_t)row * (3 * CC) + CC + h * DH + d];
    const float vv = qkv[(size_t)row * (3 * CC) + 2 * CC + h * DH + d];
    Kp[((size_t)bh * NTOK + t) * DH + d] = (__bf16)kv;
    Vt[((size_t)bh * DH + d) * NTOK + t] = (__bf16)vv;
}

// ------------------------------ WMMA GEMM ----------------------------------
// Out[M,N] = act(A[M,K] @ W[K,N] + bias), W packed bf16 (Wp[n][k]).
// Block = 4 waves; block tile 16(M) x 256(N); each wave 16x64 (4 WMMAs/K-step).
// Whole 16xK A-strip staged in LDS once; K loop is barrier-free; all 4 B
// fragments issued before the WMMA group so loads overlap matrix ops.
template <int K, int ACT>
__global__ __launch_bounds__(128) void gemm_kernel(const float* __restrict__ A,
                                                   const __bf16* __restrict__ Wp,
                                                   const float* __restrict__ bias,
                                                   float* __restrict__ out,
                                                   int N) {
    __shared__ __attribute__((aligned(16))) __bf16 Asb[16 * K];
    const int lane = threadIdx.x & 31;
    const int wave = threadIdx.x >> 5;
    const int m0 = blockIdx.x * 16;
    const int nb = blockIdx.y * 256 + wave * 64;
    const int lo = lane & 15;
    const int hi = lane >> 4;

    // ---- stage the whole 16xK A strip as bf16 (single barrier) ----
#pragma unroll
    for (int i = threadIdx.x * 4; i < 16 * K; i += 128 * 4) {
        const int row = i / K;
        const int col = i - row * K;
        const float4 v = *(const float4*)&A[(size_t)(m0 + row) * K + col];
        __bf16* dst = Asb + i;
        dst[0] = (__bf16)v.x; dst[1] = (__bf16)v.y;
        dst[2] = (__bf16)v.z; dst[3] = (__bf16)v.w;
    }
    __syncthreads();

    v8f acc0 = {}, acc1 = {}, acc2 = {}, acc3 = {};
    const __bf16* wbase = Wp + (size_t)(nb + lo) * K + hi * 16;
    for (int k0 = 0; k0 < K; k0 += 32) {
        if (k0 + 32 < K)
            __builtin_prefetch(wbase + k0 + 32, 0, 3);
        // A frag from LDS: lanes<16 -> K{0..15}; lanes>=16 -> K{16..31}
        bf16x16 af;
        {
            const __bf16* ar = Asb + lo * K + k0 + hi * 16;
#pragma unroll
            for (int j = 0; j < 8; ++j) af[j]     = ar[j];
#pragma unroll
            for (int j = 0; j < 8; ++j) af[8 + j] = ar[8 + j];
        }
        // all 4 B frags in flight before the WMMA group
        bf16x16 b0, b1, b2, b3;
        {
            const __bf16* wr = wbase + k0;
#pragma unroll
            for (int j = 0; j < 16; ++j) b0[j] = wr[j];
            wr += (size_t)16 * K;
#pragma unroll
            for (int j = 0; j < 16; ++j) b1[j] = wr[j];
            wr += (size_t)16 * K;
#pragma unroll
            for (int j = 0; j < 16; ++j) b2[j] = wr[j];
            wr += (size_t)16 * K;
#pragma unroll
            for (int j = 0; j < 16; ++j) b3[j] = wr[j];
        }
        acc0 = __builtin_amdgcn_wmma_f32_16x16x32_bf16(false, af, false, b0, (short)0, acc0, false, false);
        acc1 = __builtin_amdgcn_wmma_f32_16x16x32_bf16(false, af, false, b1, (short)0, acc1, false, false);
        acc2 = __builtin_amdgcn_wmma_f32_16x16x32_bf16(false, af, false, b2, (short)0, acc2, false, false);
        acc3 = __builtin_amdgcn_wmma_f32_16x16x32_bf16(false, af, false, b3, (short)0, acc3, false, false);
    }
    const int rbase = m0 + hi * 8;
#pragma unroll
    for (int u = 0; u < 4; ++u) {
        const int col = nb + u * 16 + lo;
        const float bv = bias ? bias[col] : 0.0f;
        const v8f acc = (u == 0) ? acc0 : (u == 1) ? acc1 : (u == 2) ? acc2 : acc3;
#pragma unroll
        for (int i = 0; i < 8; ++i) {
            float v = acc[i] + bv;
            if (ACT == 1) v = gelu_exact(v);
            out[(size_t)(rbase + i) * N + col] = v;
        }
    }
}

// --------------------------- Flash attention -------------------------------
// q from qkv (fp32); K/V from packed bf16. out: (B*N, C).
// One wave per 16-query tile per (b, head); key tiles of 32; online softmax.
__global__ __launch_bounds__(128) void flash_attn_kernel(const float* __restrict__ qkv,
                                                         const __bf16* __restrict__ Kp,
                                                         const __bf16* __restrict__ Vt,
                                                         float* __restrict__ out) {
    __shared__ __attribute__((aligned(16))) __bf16 pbuf[4][16 * 32];
    const int lane = threadIdx.x & 31;
    const int wave = threadIdx.x >> 5;
    const int qtile = blockIdx.x * 4 + wave;
    const int h = blockIdx.y;
    const int b = blockIdx.z;
    const int q0 = qtile * 16;
    const int lo = lane & 15;
    const int hi = lane >> 4;
    const int bh = b * NHEAD + h;
    const float scale = 0.17677669529663687f;     // 1/sqrt(32)
    __bf16* P = pbuf[wave];

    // q A-frag (scaled), K-dim = dh = 32
    bf16x16 qf;
    {
        const float* qrow = qkv + (size_t)(b * NTOK + q0 + lo) * (3 * CC) + h * DH;
        const int kb = hi * 16;
#pragma unroll
        for (int j = 0; j < 8; ++j) qf[j]     = (__bf16)(qrow[kb + j] * scale);
#pragma unroll
        for (int j = 0; j < 8; ++j) qf[8 + j] = (__bf16)(qrow[8 + kb + j] * scale);
    }

    float m[8], l[8];
    v8f o0 = {}, o1 = {};
#pragma unroll
    for (int i = 0; i < 8; ++i) { m[i] = -3.0e38f; l[i] = 0.f; }

    const __bf16* Kbase = Kp + (size_t)bh * NTOK * DH;
    const __bf16* Vbase = Vt + (size_t)bh * DH * NTOK;

    for (int kt = 0; kt < NTOK; kt += 32) {
        // K^T B-frags: col = key, K-dim = d; Kp rows contiguous in d
        bf16x16 kf0, kf1;
        {
            const __bf16* kr0 = Kbase + (size_t)(kt + lo) * DH + hi * 16;
            const __bf16* kr1 = kr0 + 16 * DH;
#pragma unroll
            for (int j = 0; j < 16; ++j) kf0[j] = kr0[j];
#pragma unroll
            for (int j = 0; j < 16; ++j) kf1[j] = kr1[j];
        }
        v8f zero = {};
        v8f s0 = __builtin_amdgcn_wmma_f32_16x16x32_bf16(false, qf, false, kf0,
                                                         (short)0, zero, false, false);
        v8f s1 = __builtin_amdgcn_wmma_f32_16x16x32_bf16(false, qf, false, kf1,
                                                         (short)0, zero, false, false);
        __syncthreads();
        float p0[8], p1[8];
#pragma unroll
        for (int i = 0; i < 8; ++i) {
            float tmax = fmaxf(s0[i], s1[i]);
#pragma unroll
            for (int mask = 8; mask >= 1; mask >>= 1)
                tmax = fmaxf(tmax, __shfl_xor(tmax, mask, 32));
            const float mn = fmaxf(m[i], tmax);
            const float corr = expf(m[i] - mn);
            const float e0 = expf(s0[i] - mn);
            const float e1 = expf(s1[i] - mn);
            float rs = e0 + e1;
#pragma unroll
            for (int mask = 8; mask >= 1; mask >>= 1)
                rs += __shfl_xor(rs, mask, 32);
            l[i] = l[i] * corr + rs;
            m[i] = mn;
            o0[i] *= corr;
            o1[i] *= corr;
            p0[i] = e0;
            p1[i] = e1;
        }
        // bounce P through LDS as bf16: D-layout -> A-layout
        const int rb = hi * 8;
#pragma unroll
        for (int i = 0; i < 8; ++i) {
            P[(rb + i) * 32 + lo]      = (__bf16)p0[i];
            P[(rb + i) * 32 + 16 + lo] = (__bf16)p1[i];
        }
        __syncthreads();
        bf16x16 pf;
        {
            const __bf16* prow = P + lo * 32 + hi * 16;
#pragma unroll
            for (int j = 0; j < 8; ++j) pf[j]     = prow[j];
#pragma unroll
            for (int j = 0; j < 8; ++j) pf[8 + j] = prow[8 + j];
        }
        // V B-frags: col = d, K-dim = keys; Vt rows contiguous in key
        bf16x16 vf0, vf1;
        {
            const __bf16* vr0 = Vbase + (size_t)lo * NTOK + kt + hi * 16;
            const __bf16* vr1 = Vbase + (size_t)(16 + lo) * NTOK + kt + hi * 16;
#pragma unroll
            for (int j = 0; j < 16; ++j) vf0[j] = vr0[j];
#pragma unroll
            for (int j = 0; j < 16; ++j) vf1[j] = vr1[j];
        }
        o0 = __builtin_amdgcn_wmma_f32_16x16x32_bf16(false, pf, false, vf0,
                                                     (short)0, o0, false, false);
        o1 = __builtin_amdgcn_wmma_f32_16x16x32_bf16(false, pf, false, vf1,
                                                     (short)0, o1, false, false);
    }
    const int rb = hi * 8;
#pragma unroll
    for (int i = 0; i < 8; ++i) {
        const float inv = 1.0f / l[i];
        const size_t row = (size_t)b * NTOK + q0 + rb + i;
        out[row * CC + h * DH + lo]      = o0[i] * inv;
        out[row * CC + h * DH + 16 + lo] = o1[i] * inv;
    }
}

// ------------------------------- Launcher ----------------------------------
extern "C" void kernel_launch(void* const* d_in, const int* in_sizes, int n_in,
                              void* d_out, int out_size, void* d_ws, size_t ws_size,
                              hipStream_t stream) {
    const float* x      = (const float*)d_in[0];
    const float* pos_w  = (const float*)d_in[1];
    const float* pos_b  = (const float*)d_in[2];
    const float* n1_g   = (const float*)d_in[3];
    const float* n1_b   = (const float*)d_in[4];
    const float* qkv_w  = (const float*)d_in[5];
    const float* proj_w = (const float*)d_in[6];
    const float* proj_b = (const float*)d_in[7];
    const float* lepe_w = (const float*)d_in[8];
    const float* lepe_b = (const float*)d_in[9];
    const float* n2_g   = (const float*)d_in[10];
    const float* n2_b   = (const float*)d_in[11];
    const float* p1_w   = (const float*)d_in[12];
    const float* p1_b   = (const float*)d_in[13];
    const float* p2_w   = (const float*)d_in[14];
    const float* p2_b   = (const float*)d_in[15];
    const float* g_w    = (const float*)d_in[16];
    const float* g_b    = (const float*)d_in[17];
    float* out = (float*)d_out;

    const size_t S = (size_t)BN * CC;          // 1,179,648 floats per slab
    float* ws = (float*)d_ws;
    float* Hbuf  = ws + 0 * S;                 // h
    float* LNb   = ws + 1 * S;                 // LN scratch
    float* QKV   = ws + 2 * S;                 // (BN x 768), 3 slabs
    float* Attn  = ws + 5 * S;
    float* Lepe  = ws + 6 * S;
    float* Tb    = ws + 7 * S;
    float* Pb    = ws + 8 * S;
    float* X1    = ws + 9 * S;
    float* X2    = ws + 10 * S;
    float* Hg    = ws + 11 * S;                // (BN x 1024), 4 slabs
    float* U1    = ws + 15 * S;
    float* U2    = ws + 16 * S;
    // bf16 pack region after the fp32 slabs
    __bf16* bp = (__bf16*)(ws + 17 * S);
    __bf16* WqkvP  = bp;                           bp += (size_t)CC * 3 * CC;   // 196608
    __bf16* WprojP = bp;                           bp += (size_t)CC * CC;       // 65536
    __bf16* Wp1P   = bp;                           bp += (size_t)CC * HID;      // 262144
    __bf16* Wp2P   = bp;                           bp += (size_t)HID * CC;      // 262144
    __bf16* WgP    = bp;                           bp += (size_t)HID * CC;      // 262144
    __bf16* KpB    = bp;                           bp += (size_t)BN * CC;       // 1179648
    __bf16* VtB    = bp;

    const int EW = BN * CC / 256;              // elementwise grid = 4608
    const dim3 gQKV(BN / 16, (3 * CC) / 256);  // (288, 3)
    const dim3 gPRJ(BN / 16, CC / 256);        // (288, 1)
    const dim3 gP1(BN / 16, HID / 256);        // (288, 4)
    const dim3 gFA(NTOK / 16 / 4, NHEAD, BATCH);

    // ---- pack weights to bf16 (deterministic, every call) ----
    packw_kernel<<<(CC * 3 * CC) / 256, 256, 0, stream>>>(qkv_w, WqkvP, CC, 3 * CC);
    packw_kernel<<<(CC * CC) / 256,     256, 0, stream>>>(proj_w, WprojP, CC, CC);
    packw_kernel<<<(CC * HID) / 256,    256, 0, stream>>>(p1_w, Wp1P, CC, HID);
    packw_kernel<<<(HID * CC) / 256,    256, 0, stream>>>(p2_w, Wp2P, HID, CC);
    packw_kernel<<<(HID * CC) / 256,    256, 0, stream>>>(g_w, WgP, HID, CC);

    // h = LN(x); h = h + dwconv3(h) + pos_b
    ln_kernel<<<BN, 256, 0, stream>>>(x, n1_g, n1_b, LNb);
    dwconv_kernel<<<EW, 256, 0, stream>>>(LNb, pos_w, pos_b, LNb, Hbuf, 3);

    // ---- branch 1: a1 = attn_lepe(LN(h)); x1 = h + a1 ----
    ln_kernel<<<BN, 256, 0, stream>>>(Hbuf, n1_g, n1_b, LNb);
    gemm_kernel<CC, 0><<<gQKV, 128, 0, stream>>>(LNb, WqkvP, nullptr, QKV, 3 * CC);
    packkv_kernel<<<EW, 256, 0, stream>>>(QKV, KpB, VtB);
    flash_attn_kernel<<<gFA, 128, 0, stream>>>(QKV, KpB, VtB, Attn);
    dwconv_kernel<<<EW, 256, 0, stream>>>(LNb, lepe_w, lepe_b, nullptr, Lepe, 5);
    add2_kernel<<<EW, 256, 0, stream>>>(Attn, Lepe, Tb);
    gemm_kernel<CC, 0><<<gPRJ, 128, 0, stream>>>(Tb, WprojP, proj_b, Pb, CC);
    add2_kernel<<<EW, 256, 0, stream>>>(Hbuf, Pb, X1);

    // ---- branch 2: a2 = attn_lepe(h); x2 = LN(h + a2); x2 = x1 + x2 ----
    gemm_kernel<CC, 0><<<gQKV, 128, 0, stream>>>(Hbuf, WqkvP, nullptr, QKV, 3 * CC);
    packkv_kernel<<<EW, 256, 0, stream>>>(QKV, KpB, VtB);
    flash_attn_kernel<<<gFA, 128, 0, stream>>>(QKV, KpB, VtB, Attn);
    dwconv_kernel<<<EW, 256, 0, stream>>>(Hbuf, lepe_w, lepe_b, nullptr, Lepe, 5);
    add2_kernel<<<EW, 256, 0, stream>>>(Attn, Lepe, Tb);
    gemm_kernel<CC, 0><<<gPRJ, 128, 0, stream>>>(Tb, WprojP, proj_b, Pb, CC);
    add2_kernel<<<EW, 256, 0, stream>>>(Hbuf, Pb, Tb);
    ln_kernel<<<BN, 256, 0, stream>>>(Tb, n1_g, n1_b, LNb);
    add2_kernel<<<EW, 256, 0, stream>>>(X1, LNb, X2);

    // ---- gated MLP + final LN ----
    gemm_kernel<CC, 1><<<gP1, 128, 0, stream>>>(X2, Wp1P, p1_b, Hg, HID);
    gemm_kernel<HID, 0><<<gPRJ, 128, 0, stream>>>(Hg, Wp2P, p2_b, U1, CC);
    gemm_kernel<HID, 0><<<gPRJ, 128, 0, stream>>>(Hg, WgP, g_b, U2, CC);
    combine_kernel<<<EW, 256, 0, stream>>>(X2, U1, U2, Tb);
    ln_kernel<<<BN, 256, 0, stream>>>(Tb, n2_g, n2_b, out);
}
